// NetworkGNN_79422535237963
// MI455X (gfx1250) — matrix operand order repaired and checked by reading, not compiled, not run on hardware
//
#include <hip/hip_runtime.h>

typedef __attribute__((ext_vector_type(16))) _Float16 v16h;
typedef __attribute__((ext_vector_type(8)))  float    v8f;
typedef __attribute__((ext_vector_type(4)))  float    vf4;

#define NN  100000    // nodes
#define NE  1600000   // edges
#define DIN 128
#define DH  64
#define DC  40

static inline int cdiv_i(long a, int b) { return (int)((a + b - 1) / b); }

// ---------------------------------------------------------------------------
// CSR build (one time): count -> scan -> rowptr/cursor -> perm
// ---------------------------------------------------------------------------
__global__ void zero_int_kernel(int* __restrict__ p, int n) {
    long i = (long)blockIdx.x * blockDim.x + threadIdx.x;
    if (i < n) p[i] = 0;
}

__global__ void count_kernel(const int* __restrict__ dst, int* __restrict__ count, int ne) {
    long i = (long)blockIdx.x * blockDim.x + threadIdx.x;
    if (i < ne) atomicAdd(&count[dst[i]], 1);
}

// deg_isqrt straight from the in-degree counts (deg>0 -> rsqrt(deg), else 0)
__global__ void dis_kernel(const int* __restrict__ count, float* __restrict__ dis, int n) {
    long i = (long)blockIdx.x * blockDim.x + threadIdx.x;
    if (i < n) {
        int c = count[i];
        dis[i] = (c > 0) ? rsqrtf((float)c) : 0.0f;
    }
}

__global__ void edge_norm_kernel(const int* __restrict__ src, const int* __restrict__ dst,
                                 const float* __restrict__ dis, float* __restrict__ enorm, int ne) {
    long i = (long)blockIdx.x * blockDim.x + threadIdx.x;
    if (i < ne) enorm[i] = dis[src[i]] * dis[dst[i]];
}

// per-block inclusive scan (256 elems) + block totals
__global__ void __launch_bounds__(256)
scan_block_kernel(const int* __restrict__ in, int* __restrict__ tmp,
                  int* __restrict__ bsum, int n) {
    __shared__ int s[256];
    int i = blockIdx.x * 256 + threadIdx.x;
    s[threadIdx.x] = (i < n) ? in[i] : 0;
    __syncthreads();
    #pragma unroll
    for (int off = 1; off < 256; off <<= 1) {
        int add = (threadIdx.x >= off) ? s[threadIdx.x - off] : 0;
        __syncthreads();
        s[threadIdx.x] += add;
        __syncthreads();
    }
    if (i < n) tmp[i] = s[threadIdx.x];
    if (threadIdx.x == 255) bsum[blockIdx.x] = s[255];
}

// single-block inclusive scan of the block sums (nb <= 512)
__global__ void __launch_bounds__(512)
scan_bsum_kernel(int* __restrict__ bsum, int nb) {
    __shared__ int s[512];
    s[threadIdx.x] = (threadIdx.x < nb) ? bsum[threadIdx.x] : 0;
    __syncthreads();
    #pragma unroll
    for (int off = 1; off < 512; off <<= 1) {
        int add = (threadIdx.x >= off) ? s[threadIdx.x - off] : 0;
        __syncthreads();
        s[threadIdx.x] += add;
        __syncthreads();
    }
    if (threadIdx.x < nb) bsum[threadIdx.x] = s[threadIdx.x];
}

__global__ void finalize_rowptr_kernel(const int* __restrict__ count, const int* __restrict__ tmp,
                                       const int* __restrict__ bsum, int* __restrict__ rowptr,
                                       int* __restrict__ cursor, int n, int ne) {
    long i = (long)blockIdx.x * blockDim.x + threadIdx.x;
    if (i < n) {
        int b = (int)(i >> 8);
        int pre = ((b > 0) ? bsum[b - 1] : 0) + tmp[i] - count[i];   // exclusive prefix
        rowptr[i] = pre;
        cursor[i] = pre;
    }
    if (i == 0) rowptr[n] = ne;
}

__global__ void fill_perm_kernel(const int* __restrict__ dst, int* __restrict__ cursor,
                                 int* __restrict__ perm, int ne) {
    long e = (long)blockIdx.x * blockDim.x + threadIdx.x;
    if (e < ne) {
        int p = atomicAdd(&cursor[dst[e]], 1);
        perm[p] = (int)e;
    }
}

// ---------------------------------------------------------------------------
// Fused per-node gather + norm scale + segment-sum + ReLU (no fp atomics).
// 16 threads per node, each owning a float4 slice of the 64 features.
// ---------------------------------------------------------------------------
__global__ void __launch_bounds__(256)
gather_agg_relu_kernel(const float* __restrict__ t, const int* __restrict__ rowptr,
                       const int* __restrict__ perm, const int* __restrict__ src,
                       const float* __restrict__ enorm, float* __restrict__ out) {
    long tid = (long)blockIdx.x * blockDim.x + threadIdx.x;
    int n = (int)(tid >> 4);
    if (n >= NN) return;
    int q = (int)(tid & 15);
    int beg = rowptr[n], end = rowptr[n + 1];
    vf4 acc = { 0.0f, 0.0f, 0.0f, 0.0f };
    for (int k = beg; k < end; ++k) {
        int e = perm[k];
        float nv = enorm[e];
        vf4 v = *(const vf4*)(t + (long)src[e] * DH + q * 4);
        acc += v * nv;
    }
    vf4 r;
    #pragma unroll
    for (int j = 0; j < 4; ++j) r[j] = fmaxf(acc[j], 0.0f);
    *(vf4*)(out + (long)n * DH + q * 4) = r;
}

// ---------------------------------------------------------------------------
// WMMA GEMM:  Out[M x ncols] = concat(A0,A1,A2)[M x K] * W[K x ncols] + bias
//   K = NPARTS * CPP * 32, each part has row stride lda, K-per-part = CPP*32.
//   NCOLT 16-col output tiles (cols >= ncols are zero-padded / masked).
//   B fragments pre-swizzled into LDS in this kernel's contiguous-k lane order:
//     lane l -> column (l&15), 16 consecutive K starting at (l<16 ? 0 : 16).
//   A fragment (same contiguous k order as B):
//     lane l<16 : K runs {0..15};  lane>=16 : K runs {16..31}.
// ---------------------------------------------------------------------------
template<int NPARTS, int CPP, int NCOLT>
__global__ void __launch_bounds__(256)
gemm_wmma_kernel(const float* __restrict__ a0, const float* __restrict__ a1,
                 const float* __restrict__ a2, int lda,
                 const float* __restrict__ w, const float* __restrict__ bias,
                 float* __restrict__ out, int ldo, int mtiles, int ncols) {
    constexpr int NCHUNK = NPARTS * CPP;
    __shared__ __align__(32) _Float16 bfrag[NCHUNK * NCOLT * 32 * 16];

    // ---- stage weights into LDS, pre-swizzled into B-fragment layout ----
    for (int slot = threadIdx.x; slot < NCHUNK * NCOLT * 32; slot += 256) {
        int c   = slot / (NCOLT * 32);
        int rem = slot - c * (NCOLT * 32);
        int t   = rem >> 5;
        int l   = rem & 31;
        int n   = t * 16 + (l & 15);
        int kb  = c * 32 + ((l & 16) ? 16 : 0);
        _Float16* dp = &bfrag[slot * 16];
        #pragma unroll
        for (int h = 0; h < 16; ++h) {
            float v = (n < ncols) ? w[(long)(kb + h) * ncols + n] : 0.0f;
            dp[h] = (_Float16)v;
        }
    }
    __syncthreads();

    const int wave   = threadIdx.x >> 5;
    const int lane   = threadIdx.x & 31;
    const int lhalf  = (lane & 16) ? 1 : 0;
    const int nsub   = lane & 15;
    const int nwaves = gridDim.x * 8;
    const int wid0   = blockIdx.x * 8 + wave;

    const float* parts[3] = { a0, a1, a2 };

    for (int tile = wid0; tile < mtiles; tile += nwaves) {
        const long row = (long)tile * 16 + nsub;   // lanes l and l+16 share a row

        if (tile + nwaves < mtiles)                 // global_prefetch_b8 next tile
            __builtin_prefetch(parts[0] + ((long)(tile + nwaves) * 16 + nsub) * lda, 0, 3);

        v8f acc[NCOLT];
        #pragma unroll
        for (int t = 0; t < NCOLT; ++t)
            #pragma unroll
            for (int r = 0; r < 8; ++r) acc[t][r] = 0.0f;

        #pragma unroll
        for (int c = 0; c < NCHUNK; ++c) {
            const float* ap = parts[c / CPP] + row * lda + (c % CPP) * 32;
            const vf4* r0 = (const vf4*)(ap + (lhalf ? 16 : 0));
            const vf4* r1 = (const vf4*)(ap + (lhalf ? 24 : 8));
            vf4 x0 = r0[0], x1 = r0[1], y0 = r1[0], y1 = r1[1];
            v16h a;
            #pragma unroll
            for (int j = 0; j < 4; ++j) {
                a[j]      = (_Float16)x0[j];
                a[4 + j]  = (_Float16)x1[j];
                a[8 + j]  = (_Float16)y0[j];
                a[12 + j] = (_Float16)y1[j];
            }
            #pragma unroll
            for (int t = 0; t < NCOLT; ++t) {
                const v16h* bp = (const v16h*)&bfrag[((c * NCOLT + t) * 32 + lane) * 16];
                acc[t] = __builtin_amdgcn_wmma_f32_16x16x32_f16(
                    false, a, false, *bp, (short)0, acc[t], false, false);
            }
        }

        // ---- store: C/D layout -> lane l: col (l&15), 8 rows (+8 for high lanes)
        const long rbase = (long)tile * 16 + lhalf * 8;
        #pragma unroll
        for (int t = 0; t < NCOLT; ++t) {
            int col = t * 16 + nsub;
            if (col < ncols) {
                float bv = bias[col];
                #pragma unroll
                for (int r = 0; r < 8; ++r)
                    out[(rbase + r) * ldo + col] = acc[t][r] + bv;
            }
        }
    }
}

// ---------------------------------------------------------------------------
extern "C" void kernel_launch(void* const* d_in, const int* in_sizes, int n_in,
                              void* d_out, int out_size, void* d_ws, size_t ws_size,
                              hipStream_t stream) {
    const float* x     = (const float*)d_in[0];
    const int*   ei    = (const int*)  d_in[1];
    const float* w_in  = (const float*)d_in[2];
    const float* b_in  = (const float*)d_in[3];
    const float* w_gcn = (const float*)d_in[4];   // [3,64,64]
    const float* b_gcn = (const float*)d_in[5];   // [3,64]
    const float* w_la  = (const float*)d_in[6];   // [192,64]
    const float* b_la  = (const float*)d_in[7];
    const float* w_cls = (const float*)d_in[8];   // [64,40]
    const float* b_cls = (const float*)d_in[9];
    float* out = (float*)d_out;

    const int* src = ei;
    const int* dst = ei + NE;

    // workspace layout
    float* ws    = (float*)d_ws;
    float* dis   = ws;                              // N   deg_isqrt
    float* enorm = dis + NN;                        // E
    float* h0    = enorm + NE;                      // N*64
    float* h1    = h0 + (long)NN * DH;
    float* h2    = h1 + (long)NN * DH;
    float* h3    = h2 + (long)NN * DH;
    float* tbuf  = h3 + (long)NN * DH;              // N*64 (post-GEMM / agg2)
    int*   count  = (int*)(tbuf + (long)NN * DH);   // N
    int*   tmpsc  = count + NN;                     // N
    int*   bsum   = tmpsc + NN;                     // 512
    int*   rowptr = bsum + 512;                     // N+1
    int*   cursor = rowptr + NN + 1;                // N
    int*   perm   = cursor + NN;                    // E

    const int B = 256;
    const int nscan = cdiv_i(NN, 256);              // 391 blocks (<512)

    // ---- CSR-by-dst build + edge norms (one time) ----
    zero_int_kernel      <<<cdiv_i(NN, B), B, 0, stream>>>(count, NN);
    count_kernel         <<<cdiv_i(NE, B), B, 0, stream>>>(dst, count, NE);
    dis_kernel           <<<cdiv_i(NN, B), B, 0, stream>>>(count, dis, NN);
    edge_norm_kernel     <<<cdiv_i(NE, B), B, 0, stream>>>(src, dst, dis, enorm, NE);
    scan_block_kernel    <<<nscan, 256, 0, stream>>>(count, tmpsc, bsum, NN);
    scan_bsum_kernel     <<<1, 512, 0, stream>>>(bsum, nscan);
    finalize_rowptr_kernel<<<cdiv_i(NN + 1, B), B, 0, stream>>>(count, tmpsc, bsum,
                                                                rowptr, cursor, NN, NE);
    fill_perm_kernel     <<<cdiv_i(NE, B), B, 0, stream>>>(dst, cursor, perm, NE);

    const int mtiles = NN / 16;                  // 6250 exactly
    const int gblk   = cdiv_i(mtiles, 8);        // 8 waves/block, 1 tile/wave

    // lin1: h0 = x @ w_in + b_in   (K=128)
    gemm_wmma_kernel<1, 4, 4><<<gblk, B, 0, stream>>>(
        x, x, x, DIN, w_in, b_in, h0, DH, mtiles, DH);

    // 3 GCN layers: GEMM -> fused gather/scale/sum/ReLU (no fp atomics)
    float* hin = h0;
    float* houts[3] = { h1, h2, h3 };
    for (int i = 0; i < 3; ++i) {
        gemm_wmma_kernel<1, 2, 4><<<gblk, B, 0, stream>>>(
            hin, hin, hin, DH, w_gcn + (long)i * DH * DH, b_gcn + (long)i * DH,
            tbuf, DH, mtiles, DH);
        gather_agg_relu_kernel<<<cdiv_i((long)NN * 16, B), B, 0, stream>>>(
            tbuf, rowptr, perm, src, enorm, houts[i]);
        hin = houts[i];
    }

    // la: agg2(tbuf) = concat(h1,h2,h3) @ w_la + b_la   (K=192, 3 parts)
    gemm_wmma_kernel<3, 2, 4><<<gblk, B, 0, stream>>>(
        h1, h2, h3, DH, w_la, b_la, tbuf, DH, mtiles, DH);

    // classifier: out = agg2 @ w_cls + b_cls   (K=64, 40 cols -> 3 padded tiles)
    gemm_wmma_kernel<1, 2, 3><<<gblk, B, 0, stream>>>(
        tbuf, tbuf, tbuf, DH, w_cls, b_cls, out, DC, mtiles, DC);
}
